// DipoleMomentDecoder_86973087744435
// MI455X (gfx1250) — matrix operand order repaired and checked by reading, not compiled, not run on hardware
//
#include <hip/hip_runtime.h>
#include <hip/hip_bf16.h>
#include <math.h>

// ---- problem constants (match reference) ----
#define NN 131072
#define FF 256
#define HH 128
#define GG 2048
#define MEAN_C 0.0f
#define STD_C 1.0f

typedef __bf16 bf16_t;
typedef __bf16 v16bf __attribute__((ext_vector_type(16)));
typedef float  v8f   __attribute__((ext_vector_type(8)));

union Frag32B { uint4 q[2]; v16bf v; };

// D = A(16x32 bf16) * B(32x16 bf16) + C(16x16 f32)
static __device__ __forceinline__ v8f wmma_bf16(v16bf a, v16bf b, v8f c) {
  return __builtin_amdgcn_wmma_f32_16x16x32_bf16(false, a, false, b, (short)0, c,
                                                 false, false);
}

// A-fragment from LDS, row-major [16 rows][strideHalves], K-tile kt (32 wide).
// ISA 16-bit A layout: lanes 0-15 -> M=lane, halves 0-7 = K[kb..kb+7],
// halves 8-15 = K[16+kb..16+kb+7] with kb = (lane>>4)*8.
static __device__ __forceinline__ v16bf a_frag(const bf16_t* base, int strideHalves,
                                               int kt, int lane) {
  const int m  = lane & 15;
  const int kb = (lane >> 4) * 8;
  const bf16_t* r = base + m * strideHalves + kt * 32;
  Frag32B f;
  f.q[0] = *(const uint4*)(r + kb);
  f.q[1] = *(const uint4*)(r + 16 + kb);
  return f.v;
}

// B-fragment from pre-packed global blob: [nt][kt][lane][16] bf16, contiguous.
static __device__ __forceinline__ v16bf b_frag(const bf16_t* pack, int nt, int kt,
                                               int Ktiles, int lane) {
  const bf16_t* p = pack + (((size_t)nt * Ktiles + kt) * 32 + lane) * 16;
  Frag32B f;
  f.q[0] = ((const uint4*)p)[0];
  f.q[1] = ((const uint4*)p)[1];
  return f.v;
}

static __device__ __forceinline__ v8f zero8() {
  v8f z;
#pragma unroll
  for (int j = 0; j < 8; ++j) z[j] = 0.0f;
  return z;
}

static __device__ __forceinline__ float silu_f(float x) {
  return x / (1.0f + __expf(-x));
}

// ---------------------------------------------------------------------------
// Weight pre-pack: fp32 row-major (K x O) -> bf16 WMMA B-fragment layout.
// dst[((nt*Ktiles + kt)*32 + lane)*16 + j] = W[kt*32 + (lane>>4)*8 + (j&7) + (j>>3)*16][nt*16 + (lane&15)]
// Out-of-range columns (padding) become 0.
// ---------------------------------------------------------------------------
__global__ void pack_weight_kernel(const float* __restrict__ src, bf16_t* __restrict__ dst,
                                   int K, int O, int Ntiles, int Ktiles) {
  const int idx = blockIdx.x * blockDim.x + threadIdx.x;
  const int total = Ntiles * Ktiles * 512;
  if (idx >= total) return;
  const int j    = idx & 15;
  const int lane = (idx >> 4) & 31;
  const int kt   = (idx >> 9) % Ktiles;
  const int nt   = idx / (Ktiles * 512);
  const int c = nt * 16 + (lane & 15);
  const int k = kt * 32 + (lane >> 4) * 8 + (j & 7) + (j >> 3) * 16;
  float v = 0.0f;
  if (c < O && k < K) v = src[(size_t)k * O + c];
  dst[idx] = (bf16_t)v;
}

__global__ void zero_kernel(float* __restrict__ p, int n) {
  const int i = blockIdx.x * blockDim.x + threadIdx.x;
  if (i < n) p[i] = 0.0f;
}

__global__ void finalize_kernel(const float* __restrict__ gm, float* __restrict__ out,
                                int g_count) {
  const int g = blockIdx.x * blockDim.x + threadIdx.x;
  if (g < g_count) {
    const float x = gm[g * 3 + 0];
    const float y = gm[g * 3 + 1];
    const float z = gm[g * 3 + 2];
    out[g] = (__builtin_sqrtf(x * x + y * y + z * z) - MEAN_C) / STD_C;
  }
}

// ---------------------------------------------------------------------------
// Fused per-node chain: 16 nodes / block, 4 waves (wave32).
// LDS regions are overlaid by lifetime (total 61760 B):
//   A [0,24576):  Abuf(3x16x256 bf16)  -> ybuf(16x512 bf16) -> y1buf(16x256 bf16)
//   B [24576,40960): xbuf(16x512 bf16) -> mubuf(3x16x128 bf16)
//   C [40960,53248): vecw(3x16x128 bf16)
//   D [53248,61440): x1buf(16x256 bf16)
//   E [61440,61760): w1buf(3x16 f32) + outb(16x2 f32)
// ---------------------------------------------------------------------------
__global__ __launch_bounds__(128) void fused_node_kernel(
    const float* __restrict__ pos, const float* __restrict__ scaler,
    const float* __restrict__ vec, const int* __restrict__ bidx,
    const bf16_t* __restrict__ Wg0p, const bf16_t* __restrict__ W01p,
    const float* __restrict__ b01v,  const bf16_t* __restrict__ W02p,
    const float* __restrict__ b02v,  const bf16_t* __restrict__ Wg1p,
    const bf16_t* __restrict__ W11p, const float* __restrict__ b11v,
    const bf16_t* __restrict__ W12p, const float* __restrict__ b12v,
    float* __restrict__ graph_mu) {
  __shared__ __align__(16) char smem[61760];
  bf16_t* Abuf  = (bf16_t*)smem;             // [3][16][256]
  bf16_t* ybuf  = (bf16_t*)smem;             // [16][512]
  bf16_t* y1buf = (bf16_t*)smem;             // [16][256]
  bf16_t* xbuf  = (bf16_t*)(smem + 24576);   // [16][512]
  bf16_t* mubuf = (bf16_t*)(smem + 24576);   // [3][16][128]
  bf16_t* vecw  = (bf16_t*)(smem + 40960);   // [3][16][128]
  bf16_t* x1buf = (bf16_t*)(smem + 53248);   // [16][256]
  float*  w1buf = (float*)(smem + 61440);    // [3][16]
  float*  outb  = (float*)(smem + 61632);    // [16][2]

  const int tid   = threadIdx.x;
  const int lane  = tid & 31;
  const int wave  = tid >> 5;
  const int node0 = blockIdx.x * 16;
  const int col   = lane & 15;        // C-tile column owned by this lane
  const int mb    = (lane >> 4) * 8;  // C-tile row base owned by this lane

  // ---- stage inputs into LDS (fp32 -> bf16), vectorized 16B loads ----
  for (int t = tid; t < 3 * 16 * 64; t += 128) {
    const int i  = t / (16 * 64);
    const int r  = t - i * (16 * 64);
    const int n  = r >> 6;
    const int k4 = (r & 63) * 4;
    const float4 v4 = *(const float4*)(vec + ((size_t)(node0 + n) * 3 + i) * 256 + k4);
    bf16_t* d = Abuf + (i * 16 + n) * 256 + k4;
    d[0] = (bf16_t)v4.x; d[1] = (bf16_t)v4.y; d[2] = (bf16_t)v4.z; d[3] = (bf16_t)v4.w;
  }
  for (int t = tid; t < 16 * 64; t += 128) {
    const int n  = t >> 6;
    const int k4 = (t & 63) * 4;
    const float4 v4 = *(const float4*)(scaler + (size_t)(node0 + n) * 256 + k4);
    bf16_t* d = xbuf + n * 512 + k4;
    d[0] = (bf16_t)v4.x; d[1] = (bf16_t)v4.y; d[2] = (bf16_t)v4.z; d[3] = (bf16_t)v4.w;
  }
  __syncthreads();

  // ---- stage 1: proj = vector(3x16 x 256) @ Wg0(256 x 384) ----
  // 24 n-tiles split 6/wave; the three i-tiles share each B fragment.
  for (int nt = wave * 6; nt < wave * 6 + 6; ++nt) {
    v8f a0 = zero8(), a1 = zero8(), a2 = zero8();
#pragma unroll
    for (int kt = 0; kt < 8; ++kt) {
      const v16bf b = b_frag(Wg0p, nt, kt, 8, lane);
      a0 = wmma_bf16(a_frag(Abuf + 0 * 4096, 256, kt, lane), b, a0);
      a1 = wmma_bf16(a_frag(Abuf + 1 * 4096, 256, kt, lane), b, a1);
      a2 = wmma_bf16(a_frag(Abuf + 2 * 4096, 256, kt, lane), b, a2);
    }
    if (nt + 1 < 24) __builtin_prefetch(Wg0p + (size_t)(nt + 1) * 8 * 512, 0, 1);
    if (nt < 16) {  // vec_v columns -> norm over i -> x[:,256..511]
#pragma unroll
      for (int j = 0; j < 8; ++j) {
        const float s = a0[j] * a0[j] + a1[j] * a1[j] + a2[j] * a2[j];
        xbuf[(mb + j) * 512 + 256 + nt * 16 + col] = (bf16_t)__builtin_sqrtf(s);
      }
    } else {        // vec_w columns, kept for the gate multiply
      const int h = (nt - 16) * 16 + col;
#pragma unroll
      for (int j = 0; j < 8; ++j) {
        vecw[(0 * 16 + mb + j) * 128 + h] = (bf16_t)a0[j];
        vecw[(1 * 16 + mb + j) * 128 + h] = (bf16_t)a1[j];
        vecw[(2 * 16 + mb + j) * 128 + h] = (bf16_t)a2[j];
      }
    }
  }
  __syncthreads();

  // ---- stage 2: y = silu(x(16x512) @ W01(512x512) + b01) ----
  for (int nt = wave * 8; nt < wave * 8 + 8; ++nt) {
    v8f acc = zero8();
    __builtin_prefetch(W01p + (size_t)((nt + 1) & 31) * 16 * 512, 0, 1);
#pragma unroll 4
    for (int kt = 0; kt < 16; ++kt)
      acc = wmma_bf16(a_frag(xbuf, 512, kt, lane), b_frag(W01p, nt, kt, 16, lane), acc);
    const int c = nt * 16 + col;
    const float bb = b01v[c];
#pragma unroll
    for (int j = 0; j < 8; ++j)
      ybuf[(mb + j) * 512 + c] = (bf16_t)silu_f(acc[j] + bb);
  }
  __syncthreads();

  // ---- stage 3: z = y @ W02(512x256) + b02; q = silu(z[:,:128]); mu = gate*vec_w ----
  for (int nt = wave * 4; nt < wave * 4 + 4; ++nt) {
    v8f acc = zero8();
#pragma unroll 4
    for (int kt = 0; kt < 16; ++kt)
      acc = wmma_bf16(a_frag(ybuf, 512, kt, lane), b_frag(W02p, nt, kt, 16, lane), acc);
    const int c = nt * 16 + col;
    const float bb = b02v[c];
    if (nt < 8) {  // s_out -> silu -> x1[:, 0..127]
#pragma unroll
      for (int j = 0; j < 8; ++j)
        x1buf[(mb + j) * 256 + c] = (bf16_t)silu_f(acc[j] + bb);
    } else {       // gate * vec_w -> mu (stage-4 A matrix)
      const int h = c - 128;
#pragma unroll
      for (int j = 0; j < 8; ++j) {
        const float g = acc[j] + bb;
        const int m = mb + j;
        mubuf[(0 * 16 + m) * 128 + h] = (bf16_t)(g * (float)vecw[(0 * 16 + m) * 128 + h]);
        mubuf[(1 * 16 + m) * 128 + h] = (bf16_t)(g * (float)vecw[(1 * 16 + m) * 128 + h]);
        mubuf[(2 * 16 + m) * 128 + h] = (bf16_t)(g * (float)vecw[(2 * 16 + m) * 128 + h]);
      }
    }
  }
  __syncthreads();

  // ---- stage 4: proj1 = mu(3x16 x 128) @ Wg1(128 x 144 padded; col 128 = vec_w1) ----
  for (int nt = wave; nt < 9; nt += 4) {
    v8f a0 = zero8(), a1 = zero8(), a2 = zero8();
#pragma unroll
    for (int kt = 0; kt < 4; ++kt) {
      const v16bf b = b_frag(Wg1p, nt, kt, 4, lane);
      a0 = wmma_bf16(a_frag(mubuf + 0 * 2048, 128, kt, lane), b, a0);
      a1 = wmma_bf16(a_frag(mubuf + 1 * 2048, 128, kt, lane), b, a1);
      a2 = wmma_bf16(a_frag(mubuf + 2 * 2048, 128, kt, lane), b, a2);
    }
    if (nt < 8) {  // norm over i -> x1[:, 128..255]
#pragma unroll
      for (int j = 0; j < 8; ++j) {
        const float s = a0[j] * a0[j] + a1[j] * a1[j] + a2[j] * a2[j];
        x1buf[(mb + j) * 256 + 128 + nt * 16 + col] = (bf16_t)__builtin_sqrtf(s);
      }
    } else if (col == 0) {  // column 128 = vec_w1 scalar per (node, i)
#pragma unroll
      for (int j = 0; j < 8; ++j) {
        w1buf[0 * 16 + mb + j] = a0[j];
        w1buf[1 * 16 + mb + j] = a1[j];
        w1buf[2 * 16 + mb + j] = a2[j];
      }
    }
  }
  __syncthreads();

  // ---- stage 5: y1 = silu(x1(16x256) @ W11(256x256) + b11) ----
  for (int nt = wave * 4; nt < wave * 4 + 4; ++nt) {
    v8f acc = zero8();
#pragma unroll
    for (int kt = 0; kt < 8; ++kt)
      acc = wmma_bf16(a_frag(x1buf, 256, kt, lane), b_frag(W11p, nt, kt, 8, lane), acc);
    const int c = nt * 16 + col;
    const float bb = b11v[c];
#pragma unroll
    for (int j = 0; j < 8; ++j)
      y1buf[(mb + j) * 256 + c] = (bf16_t)silu_f(acc[j] + bb);
  }
  __syncthreads();

  // ---- stage 6: o = y1 @ W12(256x16 padded) + b12; only cols 0 (q) and 1 (gate) ----
  if (wave == 0) {
    v8f acc = zero8();
#pragma unroll
    for (int kt = 0; kt < 8; ++kt)
      acc = wmma_bf16(a_frag(y1buf, 256, kt, lane), b_frag(W12p, 0, kt, 8, lane), acc);
    if (col < 2) {
      const float bb = b12v[col];
#pragma unroll
      for (int j = 0; j < 8; ++j) outb[(mb + j) * 2 + col] = acc[j] + bb;
    }
  }
  __syncthreads();

  // ---- final: node_mu = gate1*vec_w1 + q1*pos ; segment-sum into graph_mu ----
  if (tid < 48) {
    const int n = tid & 15;
    const int i = tid >> 4;
    const int node = node0 + n;
    const float q1 = outb[n * 2 + 0] * STD_C + MEAN_C;
    const float g1 = outb[n * 2 + 1];
    const float nm = g1 * w1buf[i * 16 + n] + q1 * pos[(size_t)node * 3 + i];
    atomicAdd(graph_mu + (size_t)bidx[node] * 3 + i, nm);
  }
}

// ---------------------------------------------------------------------------
// Host-side launcher (graph-capture safe: only kernel launches on `stream`).
// Workspace layout (needs ~1.16 MB):
//   [0, 24576)                : graph_mu f32[2048*3]
//   then packed bf16 weights  : Wg0p 192K | W01p 512K | W02p 256K | Wg1p 36K
//                               W11p 128K | W12p 8K
// ---------------------------------------------------------------------------
extern "C" void kernel_launch(void* const* d_in, const int* in_sizes, int n_in,
                              void* d_out, int out_size, void* d_ws, size_t ws_size,
                              hipStream_t stream) {
  (void)in_sizes; (void)n_in; (void)out_size; (void)ws_size;

  const float* pos    = (const float*)d_in[0];
  const float* scaler = (const float*)d_in[1];
  const float* vec    = (const float*)d_in[2];
  const int*   bidx   = (const int*)d_in[3];
  const float* Wg0    = (const float*)d_in[4];
  const float* W01    = (const float*)d_in[5];
  const float* b01    = (const float*)d_in[6];
  const float* W02    = (const float*)d_in[7];
  const float* b02    = (const float*)d_in[8];
  const float* Wg1    = (const float*)d_in[9];
  const float* W11    = (const float*)d_in[10];
  const float* b11    = (const float*)d_in[11];
  const float* W12    = (const float*)d_in[12];
  const float* b12    = (const float*)d_in[13];
  float* out = (float*)d_out;

  char* ws = (char*)d_ws;
  float*  graph_mu = (float*)ws;
  size_t  off = (size_t)GG * 3 * sizeof(float);            // 24576
  bf16_t* Wg0p = (bf16_t*)(ws + off); off += (size_t)24 * 8  * 512 * 2;
  bf16_t* W01p = (bf16_t*)(ws + off); off += (size_t)32 * 16 * 512 * 2;
  bf16_t* W02p = (bf16_t*)(ws + off); off += (size_t)16 * 16 * 512 * 2;
  bf16_t* Wg1p = (bf16_t*)(ws + off); off += (size_t)9  * 4  * 512 * 2;
  bf16_t* W11p = (bf16_t*)(ws + off); off += (size_t)16 * 8  * 512 * 2;
  bf16_t* W12p = (bf16_t*)(ws + off); off += (size_t)1  * 8  * 512 * 2;

  // zero the segment accumulator every launch (ws is poisoned, not re-zeroed)
  zero_kernel<<<(GG * 3 + 255) / 256, 256, 0, stream>>>(graph_mu, GG * 3);

  // pack weights fp32 -> bf16 WMMA fragment layout
  pack_weight_kernel<<<(24 * 8  * 512 + 255) / 256, 256, 0, stream>>>(Wg0, Wg0p, 256, 384, 24, 8);
  pack_weight_kernel<<<(32 * 16 * 512 + 255) / 256, 256, 0, stream>>>(W01, W01p, 512, 512, 32, 16);
  pack_weight_kernel<<<(16 * 16 * 512 + 255) / 256, 256, 0, stream>>>(W02, W02p, 512, 256, 16, 16);
  pack_weight_kernel<<<(9  * 4  * 512 + 255) / 256, 256, 0, stream>>>(Wg1, Wg1p, 128, 129, 9, 4);
  pack_weight_kernel<<<(16 * 8  * 512 + 255) / 256, 256, 0, stream>>>(W11, W11p, 256, 256, 16, 8);
  pack_weight_kernel<<<(1  * 8  * 512 + 255) / 256, 256, 0, stream>>>(W12, W12p, 256, 2, 1, 8);

  // fused per-node chain: 16 nodes per block
  fused_node_kernel<<<NN / 16, 128, 0, stream>>>(
      pos, scaler, vec, bidx, Wg0p, W01p, b01, W02p, b02, Wg1p, W11p, b11, W12p, b12,
      graph_mu);

  // graph-level norm
  finalize_kernel<<<(GG + 255) / 256, 256, 0, stream>>>(graph_mu, out, GG);
}
